// MambaLayer_6897717477705
// MI455X (gfx1250) — compile-verified, matrix operand order checked by reading, not run on hardware
//
#include <hip/hip_runtime.h>
#include <hip/hip_bf16.h>
#include <math.h>

typedef __attribute__((ext_vector_type(16))) _Float16 v16h;
typedef __attribute__((ext_vector_type(8)))  _Float16 v8h;
typedef __attribute__((ext_vector_type(8)))  float    v8f;

#define BATCH   2
#define SEQLEN  2048
#define DIM     512
#define D_INNER 1024
#define D_STATE 64
#define DT_RANK 32
#define D_CONV  4
#define NROWS   (BATCH * SEQLEN)        /* 4096 token rows */
#define NXZ     (2 * D_INNER)           /* 2048 */
#define NXDBL   (DT_RANK + 2 * D_STATE) /* 160 */
#define NXDBL_P 192                     /* padded to a multiple of 64 */

/* ---------------- f32 -> f16 conversion (weights / operands) ------------- */
__global__ void f32_to_f16_kernel(const float* __restrict__ in,
                                  _Float16* __restrict__ out, int n) {
  int i = blockIdx.x * blockDim.x + threadIdx.x;
  if (i < n) out[i] = (_Float16)in[i];
}

__global__ void fill_f16_zero_kernel(_Float16* __restrict__ out, int n) {
  int i = blockIdx.x * blockDim.x + threadIdx.x;
  if (i < n) out[i] = (_Float16)0.0f;
}

/* ---------------- LayerNorm: (4096,512) f32 -> f16 ----------------------- */
__global__ void layernorm_kernel(const float* __restrict__ x,
                                 const float* __restrict__ w,
                                 const float* __restrict__ b,
                                 _Float16* __restrict__ out) {
  __shared__ float s1[256], s2[256];
  int row = blockIdx.x;
  int t = threadIdx.x;
  const float* xr = x + (size_t)row * DIM;
  float a0 = xr[t], a1 = xr[t + 256];
  s1[t] = a0 + a1;
  s2[t] = a0 * a0 + a1 * a1;
  __syncthreads();
  for (int s = 128; s > 0; s >>= 1) {
    if (t < s) { s1[t] += s1[t + s]; s2[t] += s2[t + s]; }
    __syncthreads();
  }
  float mu  = s1[0] * (1.0f / DIM);
  float var = s2[0] * (1.0f / DIM) - mu * mu;
  float inv = rsqrtf(var + 1e-5f);
  _Float16* orow = out + (size_t)row * DIM;
  orow[t]       = (_Float16)((a0 - mu) * inv * w[t] + b[t]);
  orow[t + 256] = (_Float16)((a1 - mu) * inv * w[t + 256] + b[t + 256]);
}

/* ---------------- WMMA helpers ------------------------------------------- */
__device__ __forceinline__ v16h load16(const _Float16* p) {
  /* two 16B loads: K-halves per the 16-bit A/B fragment layout */
  v8h lo = *(const v8h*)p;
  v8h hi = *(const v8h*)(p + 16);
  v16h r;
#pragma unroll
  for (int i = 0; i < 8; ++i) { r[i] = lo[i]; r[8 + i] = hi[i]; }
  return r;
}

__device__ __forceinline__ v8f wmma16(v16h a, v16h b, v8f c) {
  return __builtin_amdgcn_wmma_f32_16x16x32_f16(
      /*neg_a=*/false, a, /*neg_b=*/false, b,
      /*c_mod=*/(short)0, c, /*reuse_a=*/false, /*reuse_b=*/false);
}

/* TN GEMM: C[m,n] = sum_k A[m,k]*Bw[n,k]; A:(M,K) f16, Bw:(N,K) f16, C f32.
   One wave -> 16(M) x 64(N) strip, 4 accumulators, A fragment reused.
   Software-pipelined: next k-step's fragments are loaded into a rotated
   register set before the current 4 WMMAs issue, so loads overlap compute.
   Requires N % 64 == 0, K % 32 == 0, M % 16 == 0 (caller pads).
   MODE 0: plain store.  MODE 1: softplus(C + bias[n]).                      */
template <int MODE>
__global__ void gemm_f16_kernel(const _Float16* __restrict__ A,
                                const _Float16* __restrict__ Bw,
                                float* __restrict__ C,
                                const float* __restrict__ bias,
                                int M, int N, int K) {
  int lane   = threadIdx.x & 31;
  int wid    = blockIdx.x * (blockDim.x >> 5) + (threadIdx.x >> 5);
  int Mtiles = M >> 4;
  int Nsuper = N >> 6;
  if (wid >= Mtiles * Nsuper) return;
  int m0 = (wid % Mtiles) << 4;
  int n0 = (wid / Mtiles) << 6;

  int half  = lane >> 4;   /* fragment half per ISA layout */
  int idx16 = lane & 15;
  int kb    = half << 3;   /* 0 or 8 */

  const _Float16* arow = A  + (size_t)(m0 + idx16) * K + kb;
  const _Float16* br0  = Bw + (size_t)(n0 + idx16) * K + kb;
  const _Float16* br1  = br0 + (size_t)16 * K;
  const _Float16* br2  = br0 + (size_t)32 * K;
  const _Float16* br3  = br0 + (size_t)48 * K;

  v8f acc0 = {}, acc1 = {}, acc2 = {}, acc3 = {};

  /* prologue loads */
  v16h af = load16(arow);
  v16h b0 = load16(br0);
  v16h b1 = load16(br1);
  v16h b2 = load16(br2);
  v16h b3 = load16(br3);

  for (int k = 32; k < K; k += 32) {
    /* prefetch next k-step into fresh registers */
    v16h afn = load16(arow + k);
    v16h b0n = load16(br0 + k);
    v16h b1n = load16(br1 + k);
    v16h b2n = load16(br2 + k);
    v16h b3n = load16(br3 + k);
    /* compute on previously-loaded fragments */
    acc0 = wmma16(af, b0, acc0);
    acc1 = wmma16(af, b1, acc1);
    acc2 = wmma16(af, b2, acc2);
    acc3 = wmma16(af, b3, acc3);
    af = afn; b0 = b0n; b1 = b1n; b2 = b2n; b3 = b3n;
  }
  /* epilogue compute */
  acc0 = wmma16(af, b0, acc0);
  acc1 = wmma16(af, b1, acc1);
  acc2 = wmma16(af, b2, acc2);
  acc3 = wmma16(af, b3, acc3);

  int rbase = m0 + (half << 3); /* C/D layout: vgpr r -> row r (+8 hi half) */
  auto store_acc = [&](const v8f& acc, int nc) {
    int col = nc + idx16;
#pragma unroll
    for (int r = 0; r < 8; ++r) {
      float v = acc[r];
      if (MODE == 1) {
        v += bias[col];
        v = (v > 20.0f) ? v : log1pf(expf(v)); /* softplus */
      }
      C[(size_t)(rbase + r) * N + col] = v;
    }
  };
  store_acc(acc0, n0);
  store_acc(acc1, n0 + 16);
  store_acc(acc2, n0 + 32);
  store_acc(acc3, n0 + 48);
}

/* ---------------- depthwise causal conv (width 4) + SiLU ----------------- */
__global__ void conv_silu_kernel(const float* __restrict__ xz,
                                 const float* __restrict__ cw,
                                 const float* __restrict__ cb,
                                 float* __restrict__ xs32,
                                 _Float16* __restrict__ xs16) {
  int idx = blockIdx.x * blockDim.x + threadIdx.x;
  if (idx >= NROWS * D_INNER) return;
  int d = idx & (D_INNER - 1);
  int m = idx >> 10;            /* flat (b,l) row */
  int l = m & (SEQLEN - 1);
  float acc = cb[d];
#pragma unroll
  for (int j = 0; j < D_CONV; ++j) {
    int ll = l - (D_CONV - 1) + j;
    if (ll >= 0)
      acc += cw[d * D_CONV + j] * xz[(size_t)(m - (D_CONV - 1) + j) * NXZ + d];
  }
  float s = acc / (1.0f + expf(-acc)); /* silu */
  xs32[idx] = s;
  xs16[idx] = (_Float16)s;
}

/* ---------------- extract dt columns of x_dbl as f16 --------------------- */
__global__ void dtcvt_kernel(const float* __restrict__ xdbl,
                             _Float16* __restrict__ dt16) {
  int idx = blockIdx.x * blockDim.x + threadIdx.x;
  if (idx >= NROWS * DT_RANK) return;
  int m = idx >> 5, r = idx & 31;
  dt16[idx] = (_Float16)xdbl[(size_t)m * NXDBL_P + r];
}

/* ---------------- selective scan (sequential over L) ---------------------
   Block = 256 threads = 128 channels x 2 threads (32 states each, in VGPRs).
   B/C rows staged in LDS once per timestep; pair-sum over states via shfl.
   Fuses D-skip, z-gate (silu) and f16 conversion for the output GEMM.      */
__global__ void scan_kernel(const float* __restrict__ xdbl,
                            const float* __restrict__ delta,
                            const float* __restrict__ xs32,
                            const float* __restrict__ xz,
                            const float* __restrict__ A_log,
                            const float* __restrict__ Dskip,
                            _Float16* __restrict__ y16) {
  __shared__ float Bsh[D_STATE];
  __shared__ float Csh[D_STATE];
  int bb   = blockIdx.x >> 3;         /* batch */
  int cblk = blockIdx.x & 7;          /* 128-channel block */
  int t = threadIdx.x;
  int d = cblk * 128 + (t >> 1);
  int half = t & 1;
  int s0 = half << 5;

  float a[32], h[32];
#pragma unroll
  for (int i = 0; i < 32; ++i) {
    a[i] = -expf(A_log[(size_t)d * D_STATE + s0 + i]);
    h[i] = 0.0f;
  }
  float dsk = Dskip[d];

  for (int l = 0; l < SEQLEN; ++l) {
    size_t row = (size_t)bb * SEQLEN + l;
    const float* xr = xdbl + row * NXDBL_P;
    if (t < 64)       Bsh[t]      = xr[DT_RANK + t];
    else if (t < 128) Csh[t - 64] = xr[DT_RANK + D_STATE + (t - 64)];
    __syncthreads();

    float dt = delta[row * D_INNER + d];
    float xt = xs32[row * D_INNER + d];
    float dx = dt * xt;
    float y = 0.0f;
#pragma unroll
    for (int i = 0; i < 32; ++i) {
      float dA = expf(dt * a[i]);
      h[i] = dA * h[i] + dx * Bsh[s0 + i];
      y += h[i] * Csh[s0 + i];
    }
    y += __shfl_xor(y, 1, 32);        /* combine the two state halves */
    if (half == 0) {
      float z = xz[row * NXZ + D_INNER + d];
      float g = z / (1.0f + expf(-z));
      y16[row * D_INNER + d] = (_Float16)((y + xt * dsk) * g);
    }
    __syncthreads();
  }
}

/* ---------------- launcher ----------------------------------------------- */
extern "C" void kernel_launch(void* const* d_in, const int* in_sizes, int n_in,
                              void* d_out, int out_size, void* d_ws, size_t ws_size,
                              hipStream_t stream) {
  (void)in_sizes; (void)n_in; (void)out_size; (void)ws_size;
  const float* x      = (const float*)d_in[0];
  const float* ln_w   = (const float*)d_in[1];
  const float* ln_b   = (const float*)d_in[2];
  const float* W_in   = (const float*)d_in[3];
  const float* conv_w = (const float*)d_in[4];
  const float* conv_b = (const float*)d_in[5];
  const float* W_x    = (const float*)d_in[6];
  const float* W_dt   = (const float*)d_in[7];
  const float* b_dt   = (const float*)d_in[8];
  const float* A_log  = (const float*)d_in[9];
  const float* D_skip = (const float*)d_in[10];
  const float* W_out  = (const float*)d_in[11];
  float* out = (float*)d_out;

  char* base = (char*)d_ws;
  size_t off = 0;
  auto nb = [&](size_t bytes) -> void* {
    void* p = base + off;
    off += (bytes + 255) & ~(size_t)255;
    return p;
  };
  _Float16* w_in16  = (_Float16*)nb((size_t)NXZ * DIM * 2);
  _Float16* w_x16   = (_Float16*)nb((size_t)NXDBL_P * D_INNER * 2); /* padded */
  _Float16* w_dt16  = (_Float16*)nb((size_t)D_INNER * DT_RANK * 2);
  _Float16* w_out16 = (_Float16*)nb((size_t)DIM * D_INNER * 2);
  _Float16* xn16    = (_Float16*)nb((size_t)NROWS * DIM * 2);
  float*    xzbuf   = (float*)   nb((size_t)NROWS * NXZ * 4);
  float*    xs32    = (float*)   nb((size_t)NROWS * D_INNER * 4);
  _Float16* xs16    = (_Float16*)nb((size_t)NROWS * D_INNER * 2);
  float*    xdbl    = (float*)   nb((size_t)NROWS * NXDBL_P * 4);  /* padded */
  _Float16* dt16    = (_Float16*)nb((size_t)NROWS * DT_RANK * 2);
  float*    deltab  = (float*)   nb((size_t)NROWS * D_INNER * 4);
  _Float16* y16     = (_Float16*)nb((size_t)NROWS * D_INNER * 2);

  auto cvt = [&](const float* src, _Float16* dst, int n) {
    f32_to_f16_kernel<<<(n + 255) / 256, 256, 0, stream>>>(src, dst, n);
  };
  cvt(W_in,  w_in16,  NXZ * DIM);
  cvt(W_x,   w_x16,   NXDBL * D_INNER);
  /* zero-fill the 32 pad rows of W_x so GEMM2 runs with N=192 branch-free */
  fill_f16_zero_kernel<<<((NXDBL_P - NXDBL) * D_INNER + 255) / 256, 256, 0, stream>>>(
      w_x16 + (size_t)NXDBL * D_INNER, (NXDBL_P - NXDBL) * D_INNER);
  cvt(W_dt,  w_dt16,  D_INNER * DT_RANK);
  cvt(W_out, w_out16, DIM * D_INNER);

  layernorm_kernel<<<NROWS, 256, 0, stream>>>(x, ln_w, ln_b, xn16);

  auto gemm0 = [&](const _Float16* A, const _Float16* Bw, float* Cc,
                   int M, int N, int K) {
    int waves = (M >> 4) * (N >> 6);
    gemm_f16_kernel<0><<<(waves + 7) / 8, 256, 0, stream>>>(A, Bw, Cc, nullptr, M, N, K);
  };

  /* xz = xn @ W_in^T  (4096x2048x512) */
  gemm0(xn16, w_in16, xzbuf, NROWS, NXZ, DIM);
  /* depthwise conv + silu */
  conv_silu_kernel<<<(NROWS * D_INNER + 255) / 256, 256, 0, stream>>>(
      xzbuf, conv_w, conv_b, xs32, xs16);
  /* x_dbl = xs @ W_x^T  (4096x192x1024, 32 pad cols) */
  gemm0(xs16, w_x16, xdbl, NROWS, NXDBL_P, D_INNER);
  dtcvt_kernel<<<(NROWS * DT_RANK + 255) / 256, 256, 0, stream>>>(xdbl, dt16);
  /* delta = softplus(dt @ W_dt^T + b_dt)  (4096x1024x32) */
  {
    int waves = (NROWS >> 4) * (D_INNER >> 6);
    gemm_f16_kernel<1><<<(waves + 7) / 8, 256, 0, stream>>>(
        dt16, w_dt16, deltab, b_dt, NROWS, D_INNER, DT_RANK);
  }
  /* selective scan + D-skip + z-gate -> y16 */
  scan_kernel<<<BATCH * (D_INNER / 128), 256, 0, stream>>>(
      xdbl, deltab, xs32, xzbuf, A_log, D_skip, y16);
  /* out = y @ W_out^T  (4096x512x1024) */
  gemm0(y16, w_out16, out, NROWS, DIM, D_INNER);
}
